// ClassicalSelfAttention_65481071396770
// MI455X (gfx1250) — matrix operand order repaired and checked by reading, not compiled, not run on hardware
//
#include <hip/hip_runtime.h>
#include <hip/hip_bf16.h>

typedef __attribute__((ext_vector_type(16))) __bf16 v16bf;
typedef __attribute__((ext_vector_type(8)))  __bf16 v8bf;
typedef __attribute__((ext_vector_type(8)))  float  v8f;
typedef unsigned short u16;
typedef unsigned int   u32;
typedef __attribute__((ext_vector_type(4))) unsigned int u32x4;
typedef __attribute__((ext_vector_type(8))) int          i32x8;
typedef __attribute__((ext_vector_type(4))) int          i32x4;

#if __has_builtin(__builtin_amdgcn_tensor_load_to_lds)
#define USE_TDM 1
#else
#define USE_TDM 0
#endif

static __device__ __forceinline__ u16 f2bf(float f) {
  __hip_bfloat16 h = __float2bfloat16(f);   // round-to-nearest-even
  u16 u;
  __builtin_memcpy(&u, &h, sizeof(u));
  return u;
}

static __device__ __forceinline__ v8bf ld8(const u16* p) {
  return *reinterpret_cast<const v8bf*>(p);  // 16-byte global_load_b128
}

#if USE_TDM
// ---------------------------------------------------------------------------
// Issue a TDM 2D tile load: Global (row-major, stride_d0 elems/row, 2B elems)
// -> LDS (tile rows contiguous). D# per CDNA5 ISA.
//   group0: [1:0]=count=1 | [63:32]=lds_addr | [120:64]=global_addr | type=2
//   group1: data_size=1(2B), tensor_dim0/1, tile_dim0/1, tensor_dim0_stride
// Toolchain here is the 6-arg builtin (clang-23 / therock-10.0):
//   (uint32x4 g0, int32x8 g1, int32x4 g2, int32x4 g3, int32x8, i32 cpol)
// ---------------------------------------------------------------------------
static __device__ __forceinline__ void tdm_load_tile_2d(
    u32 lds_off, const u16* gptr, u32 tensor_d0, u32 tensor_d1,
    u32 tile_d0, u32 tile_d1, u32 stride_d0_elems) {
  unsigned long long ga = (unsigned long long)(size_t)gptr;
  u32x4 g0 = {
      1u,                                            // count=1 (valid user D#)
      lds_off,                                       // lds_addr (bytes)
      (u32)(ga & 0xFFFFFFFFu),                       // global_addr[31:0]
      (u32)((ga >> 32) & 0x01FFFFFFu) | (2u << 30)   // addr[56:32] | type=2
  };
  i32x8 g1 = {
      (int)(1u << 16),                                            // data_size=1 -> 2B
      (int)((tensor_d0 & 0xFFFFu) << 16),                         // tensor_dim0 lo16
      (int)((tensor_d0 >> 16) | ((tensor_d1 & 0xFFFFu) << 16)),   // d0 hi16 | dim1 lo16
      (int)((tensor_d1 >> 16) | (tile_d0 << 16)),                 // d1 hi16 | tile_dim0
      (int)(tile_d1),                                             // tile_dim1 (tile_dim2=0)
      (int)(stride_d0_elems),                                     // dim0_stride lo32
      0,                                                          // dim0_stride hi16 | dim1_stride lo16
      0
  };
  i32x4 z4 = {0, 0, 0, 0};
  i32x8 z8 = {0, 0, 0, 0, 0, 0, 0, 0};
  __builtin_amdgcn_tensor_load_to_lds(g0, g1, z4, z4, z8, 0);
}
#endif

// ---------------------------------------------------------------------------
// f32 -> bf16 elementwise convert (grid-stride)
// ---------------------------------------------------------------------------
__global__ __launch_bounds__(256) void cvt_bf16(const float* __restrict__ in,
                                                u16* __restrict__ out, size_t n) {
  size_t i = (size_t)blockIdx.x * blockDim.x + threadIdx.x;
  size_t stride = (size_t)gridDim.x * blockDim.x;
  for (; i < n; i += stride) out[i] = f2bf(in[i]);
}

// ---------------------------------------------------------------------------
// f32 [R,C] -> bf16 transposed [C,R], LDS-tiled 32x32 (R,C multiples of 32)
// ---------------------------------------------------------------------------
__global__ __launch_bounds__(256) void cvt_bf16_t(const float* __restrict__ in,
                                                  u16* __restrict__ out,
                                                  int R, int C) {
  __shared__ u16 tile[32][33];
  const int c0 = blockIdx.x * 32;
  const int r0 = blockIdx.y * 32;
  for (int i = threadIdx.y; i < 32; i += 8) {
    tile[i][threadIdx.x] = f2bf(in[(size_t)(r0 + i) * C + c0 + threadIdx.x]);
  }
  __syncthreads();
  for (int i = threadIdx.y; i < 32; i += 8) {
    out[(size_t)(c0 + i) * R + r0 + threadIdx.x] = tile[threadIdx.x][i];
  }
}

// ---------------------------------------------------------------------------
// WMMA GEMM:  C[M,N] = scale * (A[M,K] @ Bt[N,K]^T) + bias
//   A, Bt bf16 row-major, lda = ldb = K (K % 32 == 0).
//   Block = 256 thr = 8 waves; block tile 128(M) x 64(N); wave tile 32x32
//   (2x2 fragments -> 4 WMMA per 32-wide K-step; A frags reused x2).
//   Bt slab (64 rows x 32 k) staged to LDS by the Tensor Data Mover,
//   double-buffered: issue tile i+1, compute tile i, s_wait_tensorcnt, barrier.
// Fragment layouts used here (contiguous k, same order for A and B):
//   A: lane m=L&15, g=L>>4; VGPR0-3 <- k=g*16..+7, VGPR4-7 <- k=g*16+8..+15
//   B: lane n=L&15, g=L>>4; VGPR v <- k=g*16+2v,2v+1 (16 contiguous k / lane)
//   C: VGPR r -> M=r+8*g, N=L&15
// ---------------------------------------------------------------------------
template <bool HAS_BIAS, bool OUT_BF16, bool TRANS_STORE>
__global__ __launch_bounds__(256, 2) void gemm_bt_wmma(
    const u16* __restrict__ A, const u16* __restrict__ Bt,
    const float* __restrict__ bias, void* __restrict__ Cout,
    int M, int N, int K, float scale, int TS) {
  (void)M;
  __shared__ u16 bsl[2][64 * 32];          // two 4KB Bt slabs

  const int lane = threadIdx.x & 31;
  const int wave = threadIdx.x >> 5;
  const int wm = wave & 3;                 // 4 waves along M
  const int wn = wave >> 2;                // 2 waves along N
  const int m0 = blockIdx.y * 128 + wm * 32;
  const int nblk = blockIdx.x * 64;
  const int n0 = nblk + wn * 32;
  const int r16 = lane & 15;
  const int g   = lane >> 4;

  const u16* ap0 = A + (size_t)(m0 + r16) * K + g * 16;       // A frag rows m0..+15
  const u16* ap1 = ap0 + (size_t)16 * K;                      // A frag rows +16
  const int offb0 = (wn * 32 + r16) * 32 + g * 16;            // B frag 0 in slab
  const int offb1 = offb0 + 16 * 32;                          // B frag 1 (+16 rows)
  const u16* bsrc = Bt + (size_t)nblk * K;                    // slab row 0, k=0

  v8f acc[2][2];
#pragma unroll
  for (int fm = 0; fm < 2; ++fm)
#pragma unroll
    for (int fn = 0; fn < 2; ++fn)
      acc[fm][fn] = v8f{0.f, 0.f, 0.f, 0.f, 0.f, 0.f, 0.f, 0.f};

  const int nIter = K / 32;

#if USE_TDM
  if (wave == 0) {
    tdm_load_tile_2d((u32)(size_t)&bsl[0][0], bsrc, (u32)K, 64u, 32u, 64u, (u32)K);
    __builtin_amdgcn_s_wait_tensorcnt(0);
  }
  __syncthreads();
#endif

  for (int i = 0; i < nIter; ++i) {
#if USE_TDM
    if (wave == 0 && i + 1 < nIter) {       // DMA next slab while computing this one
      tdm_load_tile_2d((u32)(size_t)&bsl[(i + 1) & 1][0], bsrc + (i + 1) * 32,
                       (u32)K, 64u, 32u, 64u, (u32)K);
    }
    const u16* bb = &bsl[i & 1][0];
#else
    __syncthreads();
    // cooperative copy: 2048 elems, 256 threads x one 16B chunk
    {
      const int t = threadIdx.x;            // chunk id: row=t>>2, k=(t&3)*8
      const int row = t >> 2, kk = (t & 3) * 8;
      *reinterpret_cast<v8bf*>(&bsl[0][row * 32 + kk]) =
          ld8(bsrc + (size_t)row * K + i * 32 + kk);
    }
    __syncthreads();
    const u16* bb = &bsl[0][0];
#endif

    union { v16bf v; v8bf h[2]; } a0, a1;
    a0.h[0] = ld8(ap0);
    a0.h[1] = ld8(ap0 + 8);
    a1.h[0] = ld8(ap1);
    a1.h[1] = ld8(ap1 + 8);
    __builtin_prefetch(ap0 + 32, 0, 1);
    __builtin_prefetch(ap1 + 32, 0, 1);
    v16bf b0 = *reinterpret_cast<const v16bf*>(bb + offb0);   // 2x ds_load_b128
    v16bf b1 = *reinterpret_cast<const v16bf*>(bb + offb1);

    acc[0][0] = __builtin_amdgcn_wmma_f32_16x16x32_bf16(false, a0.v, false, b0,
                                                        (short)0, acc[0][0], false, false);
    acc[0][1] = __builtin_amdgcn_wmma_f32_16x16x32_bf16(false, a0.v, false, b1,
                                                        (short)0, acc[0][1], false, false);
    acc[1][0] = __builtin_amdgcn_wmma_f32_16x16x32_bf16(false, a1.v, false, b0,
                                                        (short)0, acc[1][0], false, false);
    acc[1][1] = __builtin_amdgcn_wmma_f32_16x16x32_bf16(false, a1.v, false, b1,
                                                        (short)0, acc[1][1], false, false);
    ap0 += 32;
    ap1 += 32;

#if USE_TDM
    if (wave == 0) __builtin_amdgcn_s_wait_tensorcnt(0);  // next slab landed
    __syncthreads();                                      // all done with slab i
#endif
  }

  float bv[2] = {0.f, 0.f};
  if (HAS_BIAS) {
    bv[0] = bias[n0 + r16];
    bv[1] = bias[n0 + 16 + r16];
  }
#pragma unroll
  for (int fm = 0; fm < 2; ++fm) {
#pragma unroll
    for (int fn = 0; fn < 2; ++fn) {
      const v8f cc = acc[fm][fn];
      const int n = n0 + fn * 16 + r16;
#pragma unroll
      for (int r = 0; r < 8; ++r) {
        const int m = m0 + fm * 16 + r + 8 * g;
        const float v = cc[r] * scale + bv[fn];
        if (TRANS_STORE) {
          const int bi = m / TS;                 // out[b][n][s], bf16
          const int s  = m - bi * TS;
          reinterpret_cast<u16*>(Cout)[((size_t)bi * N + n) * TS + s] = f2bf(v);
        } else if (OUT_BF16) {
          reinterpret_cast<u16*>(Cout)[(size_t)m * N + n] = f2bf(v);
        } else {
          reinterpret_cast<float*>(Cout)[(size_t)m * N + n] = v;
        }
      }
    }
  }
}

// ---------------------------------------------------------------------------
// Row softmax: f32 scores row -> bf16 probabilities. One block per row.
// ---------------------------------------------------------------------------
__global__ __launch_bounds__(256) void softmax_bf16(const float* __restrict__ Sc,
                                                    u16* __restrict__ P, int n) {
  const int row = blockIdx.x;
  const float* src = Sc + (size_t)row * n;
  const int tid = threadIdx.x;
  __shared__ float red[8];

  float m = -3.402823466e38f;
  for (int i = tid; i < n; i += 256) m = fmaxf(m, src[i]);
  for (int off = 16; off > 0; off >>= 1) m = fmaxf(m, __shfl_xor(m, off, 32));
  if ((tid & 31) == 0) red[tid >> 5] = m;
  __syncthreads();
  if (tid == 0) {
    float v = red[0];
    for (int i = 1; i < 8; ++i) v = fmaxf(v, red[i]);
    red[0] = v;
  }
  __syncthreads();
  m = red[0];
  __syncthreads();

  float s = 0.f;
  for (int i = tid; i < n; i += 256) s += __expf(src[i] - m);
  for (int off = 16; off > 0; off >>= 1) s += __shfl_xor(s, off, 32);
  if ((tid & 31) == 0) red[tid >> 5] = s;
  __syncthreads();
  if (tid == 0) {
    float v = 0.f;
    for (int i = 0; i < 8; ++i) v += red[i];
    red[0] = v;
  }
  __syncthreads();
  const float inv = 1.f / red[0];

  for (int i = tid; i < n; i += 256)
    P[(size_t)row * n + i] = f2bf(__expf(src[i] - m) * inv);
}

// ---------------------------------------------------------------------------
// Launch
// ---------------------------------------------------------------------------
extern "C" void kernel_launch(void* const* d_in, const int* in_sizes, int n_in,
                              void* d_out, int out_size, void* d_ws, size_t ws_size,
                              hipStream_t stream) {
  (void)in_sizes; (void)n_in; (void)out_size; (void)ws_size;
  constexpr int B = 4, S = 2048, D = 1024;

  const float* x  = (const float*)d_in[0];
  const float* Wq = (const float*)d_in[1];
  const float* bq = (const float*)d_in[2];
  const float* Wk = (const float*)d_in[3];
  const float* bk = (const float*)d_in[4];
  const float* Wv = (const float*)d_in[5];
  const float* bv = (const float*)d_in[6];

  char* w = (char*)d_ws;
  auto take = [&](size_t bytes) { void* p = (void*)w; w += bytes; return p; };
  u16* xb  = (u16*)take((size_t)B * S * D * 2);   // x bf16 [B*S, D]
  u16* Wqt = (u16*)take((size_t)D * D * 2);       // Wq^T bf16 [D, D]
  u16* Wkt = (u16*)take((size_t)D * D * 2);
  u16* Wvt = (u16*)take((size_t)D * D * 2);
  u16* Qb  = (u16*)take((size_t)B * S * D * 2);   // Q bf16 [B*S, D]
  u16* Kb  = (u16*)take((size_t)B * S * D * 2);   // K bf16 [B*S, D]
  u16* Vt  = (u16*)take((size_t)B * D * S * 2);   // V^T bf16 [B][D][S]
  float* Sc = (float*)take((size_t)S * S * 4);    // scores f32, reused per batch
  u16* P    = (u16*)take((size_t)S * S * 2);      // softmax probs bf16

  // 1) convert activations / weights to bf16 (weights transposed)
  cvt_bf16<<<2048, 256, 0, stream>>>(x, xb, (size_t)B * S * D);
  {
    dim3 tb(32, 8), tg(D / 32, D / 32);
    cvt_bf16_t<<<tg, tb, 0, stream>>>(Wq, Wqt, D, D);
    cvt_bf16_t<<<tg, tb, 0, stream>>>(Wk, Wkt, D, D);
    cvt_bf16_t<<<tg, tb, 0, stream>>>(Wv, Wvt, D, D);
  }

  // 2) QKV projections: [8192,1024] @ [1024,1024] + bias
  const dim3 blk(256);
  const dim3 gproj(D / 64, (B * S) / 128);
  gemm_bt_wmma<true, true, false><<<gproj, blk, 0, stream>>>(
      xb, Wqt, bq, Qb, B * S, D, D, 1.0f, 0);
  gemm_bt_wmma<true, true, false><<<gproj, blk, 0, stream>>>(
      xb, Wkt, bk, Kb, B * S, D, D, 1.0f, 0);
  gemm_bt_wmma<true, true, true><<<gproj, blk, 0, stream>>>(
      xb, Wvt, bv, Vt, B * S, D, D, 1.0f, S);  // store V transposed per batch

  // 3) attention per batch (scores buffer reused; stream serializes kernels)
  const float inv_sqrt_d = 0.03125f;  // 1/sqrt(1024)
  for (int b = 0; b < B; ++b) {
    gemm_bt_wmma<false, false, false><<<dim3(S / 64, S / 128), blk, 0, stream>>>(
        Qb + (size_t)b * S * D, Kb + (size_t)b * S * D, nullptr, Sc,
        S, S, D, inv_sqrt_d, 0);
    softmax_bf16<<<S, 256, 0, stream>>>(Sc, P, S);
    gemm_bt_wmma<false, false, false><<<dim3(D / 64, S / 128), blk, 0, stream>>>(
        P, Vt + (size_t)b * (size_t)D * S, nullptr,
        (float*)d_out + (size_t)b * S * D, S, D, S, 1.0f, 0);
  }
}
